// FactorizedAttention2D_19121194402495
// MI455X (gfx1250) — matrix operand order repaired and checked by reading, not compiled, not run on hardware
//
#include <hip/hip_runtime.h>
#include <hip/hip_bf16.h>

typedef __attribute__((ext_vector_type(16))) _Float16 v16h;
typedef __attribute__((ext_vector_type(8)))  _Float16 v8h;
typedef __attribute__((ext_vector_type(4)))  _Float16 h4;
typedef __attribute__((ext_vector_type(8)))  float    v8f;
typedef __attribute__((ext_vector_type(4)))  unsigned u32x4;
typedef __attribute__((ext_vector_type(8)))  unsigned u32x8;

// ---------------------------------------------------------------------------
// WMMA helpers (CDNA5 gfx1250, wave32). 16x16x32 f16 -> f32.
// A per-lane: lane<16 row m=lane K{0..7,16..23}; lane>=16 row m=lane-16 K{8..15,24..31}
//   -> two 16B loads from a row-major tile.
// B per-lane: column n=lane%16, K{0..7,16..23} lane<16 / K{8..15,24..31} else
//   -> two 16B loads from the TRANSPOSED (n-major) matrix.
// C/D: VGPR r, lane L -> element (m = r + (L<16?0:8), n = L%16).
// ---------------------------------------------------------------------------
static __device__ inline v16h frag_a(const _Float16* p, int ld, int lane) {
  const _Float16* q = p + (size_t)(lane & 15) * ld + ((lane & 16) ? 8 : 0);
  v8h lo = *(const v8h*)q;
  v8h hi = *(const v8h*)(q + 16);
  v16h r;
#pragma unroll
  for (int i = 0; i < 8; ++i) { r[i] = lo[i]; r[i + 8] = hi[i]; }
  return r;
}

static __device__ inline v16h frag_b(const _Float16* p, int ld, int lane) {
  const _Float16* q = p + (size_t)(lane & 15) * ld + ((lane & 16) ? 8 : 0);
  v8h lo = *(const v8h*)q;
  v8h hi = *(const v8h*)(q + 16);
  v16h r;
#pragma unroll
  for (int i = 0; i < 8; ++i) { r[i] = lo[i]; r[i + 8] = hi[i]; }
  return r;
}

static __device__ inline v8f wmma32(v16h a, v16h b, v8f c) {
  return __builtin_amdgcn_wmma_f32_16x16x32_f16(false, a, false, b, (short)0, c,
                                                false, false);
}

// ---------------------------------------------------------------------------
// CDNA5 async copy: global -> LDS, 16B per lane, tracked by ASYNCcnt.
// ---------------------------------------------------------------------------
static __device__ inline unsigned lds_off(const void* p) {
  return (unsigned)(unsigned long long)p;
}

static __device__ inline void async_copy16(unsigned lds, const _Float16* g) {
  asm volatile("global_load_async_to_lds_b128 %0, %1, off"
               :
               : "v"(lds), "v"((unsigned long long)g)
               : "memory");
}

static __device__ inline void async_wait0() {
  asm volatile("s_wait_asynccnt 0x0" ::: "memory");
}

// ---------------------------------------------------------------------------
// CDNA5 Tensor Data Mover: one contiguous tile (n8 x 8 bytes) global -> LDS.
// D# group0: [1:0]=count=1, [63:32]=lds_addr, [120:64]=global_addr, [127:126]=2
// D# group1: data_size=3 (8B units), tensor_dim0 = tile_dim0 = n8, tile_dim1=1,
//            tensor_dim1=1, tensor_dim0_stride = n8.  Tracked by TENSORcnt.
// ---------------------------------------------------------------------------
static __device__ inline void tdm_load(unsigned lds, const void* g, unsigned n8) {
  unsigned long long ga = (unsigned long long)g;
  u32x4 g0;
  g0[0] = 1u;                                     // count=1, user descriptor
  g0[1] = lds;                                    // LDS byte address
  g0[2] = (unsigned)ga;                           // global_addr[31:0]
  g0[3] = (unsigned)(ga >> 32) | 0x80000000u;     // global_addr[56:32] | type=2
  u32x8 g1;
  g1[0] = 3u << 16;                               // data_size = 8B, no multicast
  g1[1] = n8 << 16;                               // tensor_dim0[15:0]
  g1[2] = 1u << 16;                               // tensor_dim0 hi=0, tensor_dim1=1
  g1[3] = n8 << 16;                               // tile_dim0
  g1[4] = 1u;                                     // tile_dim1=1, tile_dim2=0
  g1[5] = n8;                                     // tensor_dim0_stride lo
  g1[6] = 0u;
  g1[7] = 0u;
  asm volatile("tensor_load_to_lds %0, %1" : : "s"(g0), "s"(g1) : "memory");
}

static __device__ inline void tensor_wait0() {
  __builtin_amdgcn_s_wait_tensorcnt(0);
}

// ---------------------------------------------------------------------------
// Kernel 1: LayerNorm + 2D RoPE, f32 -> f16.  One wave per token.
// ---------------------------------------------------------------------------
__global__ __launch_bounds__(256) void k_ln_rope(
    const float* __restrict__ x, const float* __restrict__ pos_x,
    const float* __restrict__ pos_y, const float* __restrict__ gamma,
    const float* __restrict__ beta, _Float16* __restrict__ xr) {
  int wave = threadIdx.x >> 5, lane = threadIdx.x & 31;
  int t = blockIdx.x * 8 + wave;
  int h = (t >> 7) & 127, w = t & 127;

  const float4* xp = (const float4*)(x + (size_t)t * 256);
  float4 a0 = xp[lane];        // d = lane*4
  float4 a1 = xp[32 + lane];   // d = 128 + lane*4

  float s  = a0.x + a0.y + a0.z + a0.w + a1.x + a1.y + a1.z + a1.w;
  float s2 = a0.x * a0.x + a0.y * a0.y + a0.z * a0.z + a0.w * a0.w +
             a1.x * a1.x + a1.y * a1.y + a1.z * a1.z + a1.w * a1.w;
#pragma unroll
  for (int off = 1; off < 32; off <<= 1) {
    s  += __shfl_xor(s, off, 32);
    s2 += __shfl_xor(s2, off, 32);
  }
  float mu  = s * (1.0f / 256.0f);
  float var = s2 * (1.0f / 256.0f) - mu * mu;
  float rs  = rsqrtf(var + 1e-5f);

  const float4* gp = (const float4*)gamma;
  const float4* bp = (const float4*)beta;
  float4 g0 = gp[lane], g1 = gp[32 + lane];
  float4 b0 = bp[lane], b1 = bp[32 + lane];
  a0.x = (a0.x - mu) * rs * g0.x + b0.x;  a0.y = (a0.y - mu) * rs * g0.y + b0.y;
  a0.z = (a0.z - mu) * rs * g0.z + b0.z;  a0.w = (a0.w - mu) * rs * g0.w + b0.w;
  a1.x = (a1.x - mu) * rs * g1.x + b1.x;  a1.y = (a1.y - mu) * rs * g1.y + b1.y;
  a1.z = (a1.z - mu) * rs * g1.z + b1.z;  a1.w = (a1.w - mu) * rs * g1.w + b1.w;

  float4 p0, p1;
  p0.x = __shfl_xor(a0.x, 16, 32); p0.y = __shfl_xor(a0.y, 16, 32);
  p0.z = __shfl_xor(a0.z, 16, 32); p0.w = __shfl_xor(a0.w, 16, 32);
  p1.x = __shfl_xor(a1.x, 16, 32); p1.y = __shfl_xor(a1.y, 16, 32);
  p1.z = __shfl_xor(a1.z, 16, 32); p1.w = __shfl_xor(a1.w, 16, 32);

  float posx = pos_x[h], posy = pos_y[w];
  bool lo = lane < 16;
  const float* ap0 = &a0.x; const float* pp0 = &p0.x;
  const float* ap1 = &a1.x; const float* pp1 = &p1.x;
  _Float16 o[8];
#pragma unroll
  for (int c = 0; c < 4; ++c) {
    int i = (lane & 15) * 4 + c;
    float fr  = __expf((float)i * -0.14391156869f); // -ln(10000)/64
    float ax  = posx * fr, ay = posy * fr;
    float cxv = __cosf(ax), sxv = __sinf(ax);
    float cyv = __cosf(ay), syv = __sinf(ay);
    float r0 = lo ? (ap0[c] * cxv - pp0[c] * sxv)
                  : (pp0[c] * sxv + ap0[c] * cxv);
    float r1 = lo ? (ap1[c] * cyv - pp1[c] * syv)
                  : (pp1[c] * syv + ap1[c] * cyv);
    o[c]     = (_Float16)r0;
    o[4 + c] = (_Float16)r1;
  }
  h4 h0 = {o[0], o[1], o[2], o[3]};
  h4 h1 = {o[4], o[5], o[6], o[7]};
  *(h4*)(xr + (size_t)t * 256 + lane * 4)       = h0;
  *(h4*)(xr + (size_t)t * 256 + 128 + lane * 4) = h1;
}

// ---------------------------------------------------------------------------
// Kernel 2: transpose + convert weights to f16 (n-major for B-operand loads).
// ---------------------------------------------------------------------------
__global__ __launch_bounds__(256) void k_prep_w(
    const float* __restrict__ wqkv, const float* __restrict__ wout,
    _Float16* __restrict__ wqkvT, _Float16* __restrict__ woT) {
  int i = blockIdx.x * 256 + threadIdx.x;
  if (i < 768 * 256) {
    int n = i >> 8, k = i & 255;
    wqkvT[n * 256 + k] = (_Float16)wqkv[k * 768 + n];
  }
  if (i < 256 * 256) {
    int n = i >> 8, k = i & 255;
    woT[n * 256 + k] = (_Float16)wout[k * 256 + n];
  }
}

// ---------------------------------------------------------------------------
// Kernel 3: QKV GEMM [131072,256]x[256,768]. Block = 128Mx128N, ASYNCcnt
// double-buffered LDS staging of A and B k-slices, scatter to head-major q/k/v.
// ---------------------------------------------------------------------------
__global__ __launch_bounds__(256) void k_qkv_gemm(
    const _Float16* __restrict__ xr, const _Float16* __restrict__ wT,
    _Float16* __restrict__ qb, _Float16* __restrict__ kb,
    _Float16* __restrict__ vb) {
  __shared__ _Float16 As[2][128 * 32];
  __shared__ _Float16 Bs[2][128 * 32];
  int tid = threadIdx.x, wave = tid >> 5, lane = tid & 31;
  int m_blk = blockIdx.x * 128, n0 = blockIdx.y * 128;

  int srow = tid >> 1, scol = (tid & 1) * 16;
  const _Float16* Ag = xr + (size_t)(m_blk + srow) * 256 + scol;
  const _Float16* Bg = wT + (size_t)(n0 + srow) * 256 + scol;
  unsigned aoff0 = lds_off(&As[0][srow * 32 + scol]);
  unsigned aoff1 = lds_off(&As[1][srow * 32 + scol]);
  unsigned boff0 = lds_off(&Bs[0][srow * 32 + scol]);
  unsigned boff1 = lds_off(&Bs[1][srow * 32 + scol]);

  async_copy16(aoff0, Ag);
  async_copy16(boff0, Bg);

  v8f zero = {};
  v8f acc[8];
#pragma unroll
  for (int i = 0; i < 8; ++i) acc[i] = zero;

  for (int ks = 0; ks < 8; ++ks) {
    async_wait0();
    __syncthreads();
    if (ks < 7) {
      async_copy16((ks & 1) ? aoff0 : aoff1, Ag + (ks + 1) * 32);
      async_copy16((ks & 1) ? boff0 : boff1, Bg + (ks + 1) * 32);
    }
    const _Float16* At = As[ks & 1];
    const _Float16* Bt = Bs[ks & 1];
    v16h a = frag_a(At + wave * 16 * 32, 32, lane);
#pragma unroll
    for (int nt = 0; nt < 8; ++nt)
      acc[nt] = wmma32(a, frag_b(Bt + nt * 16 * 32, 32, lane), acc[nt]);
  }

  int m0 = m_blk + wave * 16;
  int half8 = (lane & 16) ? 8 : 0;
#pragma unroll
  for (int nt = 0; nt < 8; ++nt) {
    int ng   = n0 + nt * 16 + (lane & 15);
    int part = ng >> 8;            // 0=q 1=k 2=v
    int nn   = ng & 255;
    int head = nn >> 5, dd = nn & 31;
    _Float16* dst = (part == 0) ? qb : ((part == 1) ? kb : vb);
#pragma unroll
    for (int r = 0; r < 8; ++r) {
      int t  = m0 + r + half8;
      int b_ = t >> 14, h = (t >> 7) & 127, w = t & 127;
      size_t idx = ((((size_t)(b_ * 8 + head) * 128 + w) * 128 + h) * 32 + dd);
      dst[idx] = (_Float16)acc[nt][r];
    }
  }
}

// ---------------------------------------------------------------------------
// Kernel 4: column attention (contract over H). One block per (b,head,w).
// Q/K/V tiles fetched by the Tensor Data Mover (wave 0 issues 3 descriptors,
// TENSORcnt-waited); V lands in the Ps region and is LDS-transposed into Vt.
// ---------------------------------------------------------------------------
__global__ __launch_bounds__(256) void k_attn_col(
    const _Float16* __restrict__ qb, const _Float16* __restrict__ kb,
    const _Float16* __restrict__ vb, _Float16* __restrict__ ox) {
  __shared__ _Float16 Qs[128 * 32];
  __shared__ _Float16 Ks[128 * 32];
  __shared__ _Float16 Vt[32 * 128];   // transposed V: [dd][H]
  __shared__ _Float16 Ps[128 * 128];  // V staging area, then softmax numerators

  int tid = threadIdx.x, wave = tid >> 5, lane = tid & 31;
  int s = blockIdx.x;           // bh*128 + w
  int bh = s >> 7, w = s & 127;
  size_t base = (size_t)s * 4096;

  if (wave == 0) {
    tdm_load(lds_off(Qs), qb + base, 1024);   // 8KB tile
    tdm_load(lds_off(Ks), kb + base, 1024);
    tdm_load(lds_off(Ps), vb + base, 1024);
    tensor_wait0();
  }
  __syncthreads();
#pragma unroll
  for (int i = tid; i < 4096; i += 256)
    Vt[(i & 31) * 128 + (i >> 5)] = Ps[i];
  __syncthreads();

  int m0 = wave * 16;
  v16h aq = frag_a(Qs + m0 * 32, 32, lane);
  v8f sc[8];
#pragma unroll
  for (int nt = 0; nt < 8; ++nt) {
    v8f z = {};
    v16h bk = frag_b(Ks + nt * 16 * 32, 32, lane);
    sc[nt] = wmma32(aq, bk, z);
  }

  const float scale = 0.17677669529663689f;  // 1/sqrt(32)
  float rsum[8];
#pragma unroll
  for (int r = 0; r < 8; ++r) {
    float mx = -3.0e38f;
#pragma unroll
    for (int nt = 0; nt < 8; ++nt) mx = fmaxf(mx, sc[nt][r]);
#pragma unroll
    for (int off = 1; off < 16; off <<= 1) mx = fmaxf(mx, __shfl_xor(mx, off, 32));
    float sum = 0.f;
#pragma unroll
    for (int nt = 0; nt < 8; ++nt) {
      float e = __expf((sc[nt][r] - mx) * scale);
      sc[nt][r] = e;
      sum += e;
    }
#pragma unroll
    for (int off = 1; off < 16; off <<= 1) sum += __shfl_xor(sum, off, 32);
    rsum[r] = sum;
  }

  int mrow = m0 + ((lane & 16) ? 8 : 0);
  __syncthreads();  // everyone done with the V staging image in Ps
#pragma unroll
  for (int nt = 0; nt < 8; ++nt) {
    int col = nt * 16 + (lane & 15);
#pragma unroll
    for (int r = 0; r < 8; ++r)
      Ps[(mrow + r) * 128 + col] = (_Float16)sc[nt][r];
  }
  __syncthreads();

  v8f o0 = {}, o1 = {};
#pragma unroll
  for (int k0 = 0; k0 < 128; k0 += 32) {
    v16h pa = frag_a(Ps + m0 * 128 + k0, 128, lane);
    v16h b0 = frag_b(Vt + k0, 128, lane);
    v16h b1 = frag_b(Vt + 16 * 128 + k0, 128, lane);
    o0 = wmma32(pa, b0, o0);
    o1 = wmma32(pa, b1, o1);
  }

  // out_x layout: [B*HEADS][H][W][32]
#pragma unroll
  for (int r = 0; r < 8; ++r) {
    int hrow = mrow + r;
    float inv = 1.0f / rsum[r];
    size_t ob = (((size_t)(bh * 128 + hrow) * 128) + w) * 32;
    ox[ob + (lane & 15)]      = (_Float16)(o0[r] * inv);
    ox[ob + 16 + (lane & 15)] = (_Float16)(o1[r] * inv);
  }
}

// ---------------------------------------------------------------------------
// Kernel 5: row attention (contract over W) of out_x with itself.
// One block per (b,head,h); X tile fetched via TDM. Output token-major f16.
// ---------------------------------------------------------------------------
__global__ __launch_bounds__(256) void k_attn_row(
    const _Float16* __restrict__ ox, _Float16* __restrict__ oy) {
  __shared__ _Float16 Xs[128 * 32];
  __shared__ _Float16 Xt[32 * 128];
  __shared__ _Float16 Ps[128 * 128];

  int tid = threadIdx.x, wave = tid >> 5, lane = tid & 31;
  int s = blockIdx.x;           // bh*128 + h
  int bh = s >> 7, h = s & 127;
  int b_ = bh >> 3, head = bh & 7;
  size_t base = (size_t)s * 4096;

  if (wave == 0) {
    tdm_load(lds_off(Xs), ox + base, 1024);
    tensor_wait0();
  }
  __syncthreads();
#pragma unroll
  for (int i = tid; i < 4096; i += 256)
    Xt[(i & 31) * 128 + (i >> 5)] = Xs[i];
  __syncthreads();

  int m0 = wave * 16;
  v16h aq = frag_a(Xs + m0 * 32, 32, lane);
  v8f sc[8];
#pragma unroll
  for (int nt = 0; nt < 8; ++nt) {
    v8f z = {};
    v16h bk = frag_b(Xs + nt * 16 * 32, 32, lane);
    sc[nt] = wmma32(aq, bk, z);
  }

  const float scale = 0.17677669529663689f;
  float rsum[8];
#pragma unroll
  for (int r = 0; r < 8; ++r) {
    float mx = -3.0e38f;
#pragma unroll
    for (int nt = 0; nt < 8; ++nt) mx = fmaxf(mx, sc[nt][r]);
#pragma unroll
    for (int off = 1; off < 16; off <<= 1) mx = fmaxf(mx, __shfl_xor(mx, off, 32));
    float sum = 0.f;
#pragma unroll
    for (int nt = 0; nt < 8; ++nt) {
      float e = __expf((sc[nt][r] - mx) * scale);
      sc[nt][r] = e;
      sum += e;
    }
#pragma unroll
    for (int off = 1; off < 16; off <<= 1) sum += __shfl_xor(sum, off, 32);
    rsum[r] = sum;
  }

  int mrow = m0 + ((lane & 16) ? 8 : 0);
#pragma unroll
  for (int nt = 0; nt < 8; ++nt) {
    int col = nt * 16 + (lane & 15);
#pragma unroll
    for (int r = 0; r < 8; ++r)
      Ps[(mrow + r) * 128 + col] = (_Float16)sc[nt][r];
  }
  __syncthreads();

  v8f o0 = {}, o1 = {};
#pragma unroll
  for (int k0 = 0; k0 < 128; k0 += 32) {
    v16h pa = frag_a(Ps + m0 * 128 + k0, 128, lane);
    v16h b0 = frag_b(Xt + k0, 128, lane);
    v16h b1 = frag_b(Xt + 16 * 128 + k0, 128, lane);
    o0 = wmma32(pa, b0, o0);
    o1 = wmma32(pa, b1, o1);
  }

  // out_y token-major: [b][h][w][head*32+dd]
#pragma unroll
  for (int r = 0; r < 8; ++r) {
    int wrow = mrow + r;
    float inv = 1.0f / rsum[r];
    size_t t = ((size_t)(b_ * 128 + h) * 128 + wrow);
    oy[t * 256 + head * 32 + (lane & 15)]      = (_Float16)(o0[r] * inv);
    oy[t * 256 + head * 32 + 16 + (lane & 15)] = (_Float16)(o1[r] * inv);
  }
}

// ---------------------------------------------------------------------------
// Kernel 6: output projection [131072,256]x[256,256] + bias -> f32.
// Same ASYNCcnt double-buffered staging as kernel 3.
// ---------------------------------------------------------------------------
__global__ __launch_bounds__(256) void k_out_gemm(
    const _Float16* __restrict__ oy, const _Float16* __restrict__ woT,
    const float* __restrict__ bout, float* __restrict__ out) {
  __shared__ _Float16 As[2][128 * 32];
  __shared__ _Float16 Bs[2][128 * 32];
  int tid = threadIdx.x, wave = tid >> 5, lane = tid & 31;
  int m_blk = blockIdx.x * 128, n0 = blockIdx.y * 128;

  int srow = tid >> 1, scol = (tid & 1) * 16;
  const _Float16* Ag = oy + (size_t)(m_blk + srow) * 256 + scol;
  const _Float16* Bg = woT + (size_t)(n0 + srow) * 256 + scol;
  unsigned aoff0 = lds_off(&As[0][srow * 32 + scol]);
  unsigned aoff1 = lds_off(&As[1][srow * 32 + scol]);
  unsigned boff0 = lds_off(&Bs[0][srow * 32 + scol]);
  unsigned boff1 = lds_off(&Bs[1][srow * 32 + scol]);

  async_copy16(aoff0, Ag);
  async_copy16(boff0, Bg);

  v8f zero = {};
  v8f acc[8];
#pragma unroll
  for (int i = 0; i < 8; ++i) acc[i] = zero;

  for (int ks = 0; ks < 8; ++ks) {
    async_wait0();
    __syncthreads();
    if (ks < 7) {
      async_copy16((ks & 1) ? aoff0 : aoff1, Ag + (ks + 1) * 32);
      async_copy16((ks & 1) ? boff0 : boff1, Bg + (ks + 1) * 32);
    }
    const _Float16* At = As[ks & 1];
    const _Float16* Bt = Bs[ks & 1];
    v16h a = frag_a(At + wave * 16 * 32, 32, lane);
#pragma unroll
    for (int nt = 0; nt < 8; ++nt)
      acc[nt] = wmma32(a, frag_b(Bt + nt * 16 * 32, 32, lane), acc[nt]);
  }

  int m0 = m_blk + wave * 16;
  int half8 = (lane & 16) ? 8 : 0;
#pragma unroll
  for (int nt = 0; nt < 8; ++nt) {
    int n = n0 + nt * 16 + (lane & 15);
    float bv = bout[n];
#pragma unroll
    for (int r = 0; r < 8; ++r)
      out[(size_t)(m0 + r + half8) * 256 + n] = acc[nt][r] + bv;
  }
}

// ---------------------------------------------------------------------------
extern "C" void kernel_launch(void* const* d_in, const int* in_sizes, int n_in,
                              void* d_out, int out_size, void* d_ws,
                              size_t ws_size, hipStream_t stream) {
  (void)in_sizes; (void)n_in; (void)out_size; (void)ws_size;
  const float* x     = (const float*)d_in[0];
  const float* pos_x = (const float*)d_in[1];
  const float* pos_y = (const float*)d_in[2];
  const float* gamma = (const float*)d_in[3];
  const float* beta  = (const float*)d_in[4];
  const float* wqkv  = (const float*)d_in[5];
  const float* wout  = (const float*)d_in[6];
  const float* bout  = (const float*)d_in[7];
  float* out = (float*)d_out;

  const size_t SL = (size_t)131072 * 256;  // elements per f16 tensor
  _Float16* ws    = (_Float16*)d_ws;
  _Float16* xr    = ws;            // [N,256] post LN+RoPE
  _Float16* qb    = xr + SL;       // [B*HEADS][W][H][32]
  _Float16* kb    = qb + SL;
  _Float16* vb    = kb + SL;
  _Float16* ox    = vb + SL;       // [B*HEADS][H][W][32]
  _Float16* wqkvT = ox + SL;       // [768][256]
  _Float16* woT   = wqkvT + 768 * 256;  // [256][256]
  _Float16* oy    = xr;            // reuse xr (dead after QKV GEMM)

  k_ln_rope<<<16384, 256, 0, stream>>>(x, pos_x, pos_y, gamma, beta, xr);
  k_prep_w<<<768, 256, 0, stream>>>(wqkv, wout, wqkvT, woT);
  k_qkv_gemm<<<dim3(1024, 6), 256, 0, stream>>>(xr, wqkvT, qb, kb, vb);
  k_attn_col<<<8192, 256, 0, stream>>>(qb, kb, vb, ox);
  k_attn_row<<<8192, 256, 0, stream>>>(ox, oy);
  k_out_gemm<<<dim3(1024, 2), 256, 0, stream>>>(oy, woT, bout, out);
}
